// Trend_aware_attention_27633819582588
// MI455X (gfx1250) — matrix operand order repaired and checked by reading, not compiled, not run on hardware
//
#include <hip/hip_runtime.h>
#include <hip/hip_bf16.h>

typedef __attribute__((ext_vector_type(16))) __bf16 v16bf;
typedef __attribute__((ext_vector_type(8)))  float  v8f;
typedef unsigned short ushort_t;

namespace {
constexpr int kB  = 8;     // batch
constexpr int kT  = 64;    // time steps
constexpr int kN  = 128;   // vertices
constexpr int kH  = 8;     // heads
constexpr int kDH = 64;    // head dim
constexpr int kD  = 512;   // model dim
constexpr int kKS = 3;     // causal conv taps
constexpr int kTP = kT + 2;              // padded time rows (2 zero rows in front)
constexpr long kOutRow = (long)kN * kD;  // d_out stride between consecutive t
}

__device__ __forceinline__ ushort_t f2bf(float f) {
  unsigned u = __float_as_uint(f);
  u += 0x7FFFu + ((u >> 16) & 1u);  // round-to-nearest-even
  return (ushort_t)(u >> 16);
}

union ABPack { uint4 q[2]; v16bf v; };

// A operand (16x32 bf16, row-major source, row stride ldk):
// lane&15 = row; lanes<16 take K[0..7]+K[16..23], lanes>=16 take K[8..15]+K[24..31]
__device__ __forceinline__ v16bf load_a16(const ushort_t* p, int ldk, int lane) {
  const ushort_t* q = p + (long)(lane & 15) * ldk + ((lane & 16) ? 8 : 0);
  ABPack u;
  u.q[0] = *(const uint4*)(q);
  u.q[1] = *(const uint4*)(q + 16);
  return u.v;
}

// B operand (32x16 bf16, columns stored row-major [col][k], stride ldk):
// lane&15 = column; lanes<16 take K[0..7]+K[16..23], lanes>=16 take K[8..15]+K[24..31]
__device__ __forceinline__ v16bf load_b16(const ushort_t* p, int ldk, int lane) {
  const ushort_t* q = p + (long)(lane & 15) * ldk + ((lane & 16) ? 8 : 0);
  ABPack u;
  u.q[0] = *(const uint4*)(q);
  u.q[1] = *(const uint4*)(q + 16);
  return u.v;
}

__device__ __forceinline__ v8f wmma_bf16(v16bf a, v16bf b, v8f c) {
  return __builtin_amdgcn_wmma_f32_16x16x32_bf16(false, a, false, b, (short)0, c,
                                                 false, false);
}

// Async global -> LDS (CDNA5 path, tracked by ASYNCcnt). LDS byte address is the
// low 32 bits of the generic pointer to a __shared__ object.
__device__ __forceinline__ void async_b128_to_lds(void* lds, const ushort_t* gptr) {
  const unsigned ldsa = (unsigned)(unsigned long long)lds;
  asm volatile("global_load_async_to_lds_b128 %0, %1, off"
               :: "v"(ldsa), "v"(gptr) : "memory");
}

__device__ __forceinline__ void wait_async0() {
#if __has_builtin(__builtin_amdgcn_s_wait_asynccnt)
  __builtin_amdgcn_s_wait_asynccnt(0);
#else
  asm volatile("s_wait_asynccnt 0x0" ::: "memory");
#endif
}

// ---------------- converts / layout transforms ----------------

// X [B,T,N,D] fp32 -> Xp [B,N,T+2,D] bf16 (rows 0,1 of each (b,n) are the causal pad)
__global__ void cvt_x_pad(const float* __restrict__ X, ushort_t* __restrict__ Xp) {
  const int i = blockIdx.x * 256 + threadIdx.x;  // exactly B*T*N*D threads
  const int d = i & (kD - 1);
  const int n = (i >> 9) & (kN - 1);
  const int t = (i >> 16) & (kT - 1);
  const int b = i >> 22;
  Xp[(((long)b * kN + n) * kTP + t + 2) * kD + d] = f2bf(X[i]);
}

// zero the two pad rows of every (b,n): B*N*2*D elements
__global__ void zero_pad(ushort_t* __restrict__ Xp) {
  const int i = blockIdx.x * 256 + threadIdx.x;
  const int d = i & (kD - 1);
  const int row = (i >> 9) & 1;
  const int bn = i >> 10;
  Xp[((long)bn * kTP + row) * kD + d] = 0;
}

__global__ void cvt_f32_bf16(const float* __restrict__ in, ushort_t* __restrict__ out,
                             int count) {
  const int i = blockIdx.x * 256 + threadIdx.x;
  if (i < count) out[i] = f2bf(in[i]);
}

// Wq/Wk [Dout][Din][1][KS] -> bf16 [tap][Dout][Din] (row-major over Din)
__global__ void cvt_convw(const float* __restrict__ W, ushort_t* __restrict__ out) {
  const int i = blockIdx.x * 256 + threadIdx.x;
  if (i < kD * kD * kKS) {
    const int tap  = i % kKS;
    const int din  = (i / kKS) % kD;
    const int dout = i / (kKS * kD);
    out[((long)tap * kD + dout) * kD + din] = f2bf(W[i]);
  }
}

// ---------------- causal conv (3 taps) + BatchNorm, WMMA ----------------
// One 16(t) x 16(outCh) tile per wave; fully-unrolled 48-step K loop so every
// k-offset folds into the load's immediate and the scheduler can run loads ahead.
// Padded X rows make every A load unconditional: tap row base = t0 + tap.
__global__ __launch_bounds__(256) void conv_bn_wmma(
    const ushort_t* __restrict__ Xp, const ushort_t* __restrict__ Wb,
    const float* __restrict__ bconv, const float* __restrict__ gam,
    const float* __restrict__ beta, const float* __restrict__ mean,
    const float* __restrict__ var, ushort_t* __restrict__ out) {
  const int lane = threadIdx.x & 31;
  const int tile = blockIdx.x * 8 + (threadIdx.x >> 5);
  const int col  = (tile & 31) << 4;   // 32 column tiles of 16
  const int grp  = tile >> 5;          // (b, n, t-tile)
  const int t0   = (grp & 3) << 4;
  const int n    = (grp >> 2) & (kN - 1);
  const int b    = grp >> 9;
  const int r    = lane & 15;

  const ushort_t* xbase = Xp + ((long)b * kN + n) * kTP * kD;
  v8f acc = {};
  #pragma unroll
  for (int tap = 0; tap < kKS; ++tap) {
    const ushort_t* arow = xbase + (long)(t0 + tap) * kD;   // causal pad absorbed
    const ushort_t* wrow = Wb + ((long)tap * kD + col) * kD;
    #pragma unroll
    for (int k = 0; k < kD; k += 32)
      acc = wmma_bf16(load_a16(arow + k, kD, lane),
                      load_b16(wrow + k, kD, lane), acc);
  }

  // epilogue: fold conv bias + eval-mode BN into per-channel scale/shift
  const int c = col + r;
  const float sc = gam[c] * rsqrtf(var[c] + 1e-5f);
  const float sh = beta[c] + (bconv[c] - mean[c]) * sc;
  const int mb = (lane & 16) ? 8 : 0;
  ushort_t* obase = out + (((long)b * kN + n) * kT + t0) * kD + c;  // [B,N,T,D]
  #pragma unroll
  for (int rr = 0; rr < 8; ++rr)
    obase[(long)(mb + rr) * kD] = f2bf(acc[rr] * sc + sh);
}

// ---------------- V projection: V = X @ Wv^T + bv (bf16 out, [B,N,T,D]) ----------------
__global__ __launch_bounds__(256) void vproj_wmma(
    const ushort_t* __restrict__ Xp, const ushort_t* __restrict__ Wb,
    const float* __restrict__ bias, ushort_t* __restrict__ out) {
  const int lane = threadIdx.x & 31;
  const int tile = blockIdx.x * 8 + (threadIdx.x >> 5);
  const int col  = (tile & 31) << 4;
  const int grp  = tile >> 5;
  const int t0   = (grp & 3) << 4;
  const int n    = (grp >> 2) & (kN - 1);
  const int b    = grp >> 9;
  const int r    = lane & 15;

  const ushort_t* abase = Xp + (((long)b * kN + n) * kTP + 2 + t0) * kD;
  const ushort_t* wbase = Wb + (long)col * kD;
  v8f acc = {};
  #pragma unroll
  for (int k = 0; k < kD; k += 32)
    acc = wmma_bf16(load_a16(abase + k, kD, lane),
                    load_b16(wbase + k, kD, lane), acc);

  const int c = col + r;
  const float bc = bias[c];
  const int mb = (lane & 16) ? 8 : 0;
  ushort_t* obase = out + (((long)b * kN + n) * kT + t0) * kD + c;
  #pragma unroll
  for (int rr = 0; rr < 8; ++rr)
    obase[(long)(mb + rr) * kD] = f2bf(acc[rr] + bc);
}

// ---------------- attention: one (b, n, h) 64x64x64 problem per block ----------------
// LDS: Q[64x64] K[64x64] V^T[64x64] bf16 + scores[64x64] f32 = 40 KB.
// Q/K staged via async global->LDS; output written in-place over the Q buffer.
__global__ __launch_bounds__(256) void attn_wmma(const ushort_t* Qg, const ushort_t* Kg,
                                                 const ushort_t* Vg, ushort_t* Og) {
  __shared__ __align__(16) ushort_t Qs[kT * kDH];
  __shared__ __align__(16) ushort_t Ks[kT * kDH];
  __shared__ __align__(16) ushort_t VsT[kDH * kT];
  __shared__ __align__(16) float Ss[kT * kT];

  const int tid = threadIdx.x;
  const int h = blockIdx.x & (kH - 1);
  const int n = (blockIdx.x >> 3) & (kN - 1);
  const int b = blockIdx.x >> 10;
  const long base = ((long)b * kN + n) * kT * kD + (long)h * kDH;  // [B,N,T,D]

  // 512 chunks of 8 bf16 for each of Q/K/V; V transposed through registers
  for (int c = tid; c < (kT * kDH) / 8; c += 256) {
    const int t = c >> 3;
    const int d8 = (c & 7) << 3;
    const long g = base + (long)t * kD + d8;
    async_b128_to_lds(&Qs[t * kDH + d8], Qg + g);
    async_b128_to_lds(&Ks[t * kDH + d8], Kg + g);
    __align__(16) ushort_t vt[8];
    *(uint4*)vt = *(const uint4*)(Vg + g);
    #pragma unroll
    for (int i = 0; i < 8; ++i) VsT[(d8 + i) * kT + t] = vt[i];
  }
  wait_async0();
  __syncthreads();

  const int w = tid >> 5;
  const int lane = tid & 31;
  const int r = lane & 15;
  const int mb = (lane & 16) ? 8 : 0;

  // scores = Q K^T * DH^-0.5 ; 16 tiles of 16x16, 2 per wave
  for (int tIdx = w; tIdx < 16; tIdx += 8) {
    const int qi = tIdx >> 2, pj = tIdx & 3;
    v8f acc = {};
    #pragma unroll
    for (int kk = 0; kk < kDH; kk += 32) {
      v16bf a  = load_a16(&Qs[qi * 16 * kDH + kk], kDH, lane);
      v16bf bm = load_b16(&Ks[pj * 16 * kDH + kk], kDH, lane);
      acc = wmma_bf16(a, bm, acc);
    }
    #pragma unroll
    for (int rr = 0; rr < 8; ++rr)
      Ss[(qi * 16 + mb + rr) * kT + pj * 16 + r] = acc[rr] * 0.125f;
  }
  __syncthreads();

  // row softmax (fp32), attn written as bf16 over the Q LDS tile
  if (tid < kT) {
    float mx = -3.0e38f;
    for (int p = 0; p < kT; ++p) mx = fmaxf(mx, Ss[tid * kT + p]);
    float s = 0.f;
    for (int p = 0; p < kT; ++p) s += __expf(Ss[tid * kT + p] - mx);
    const float inv = 1.f / s;
    for (int p = 0; p < kT; ++p)
      Qs[tid * kT + p] = f2bf(__expf(Ss[tid * kT + p] - mx) * inv);
  }
  __syncthreads();

  // out = attn @ V ; B operand from V^T (rows = head-dim columns)
  for (int tIdx = w; tIdx < 16; tIdx += 8) {
    const int qi = tIdx >> 2, dj = tIdx & 3;
    v8f acc = {};
    #pragma unroll
    for (int kk = 0; kk < kT; kk += 32) {
      v16bf a  = load_a16(&Qs[qi * 16 * kT + kk], kT, lane);
      v16bf bm = load_b16(&VsT[dj * 16 * kT + kk], kT, lane);
      acc = wmma_bf16(a, bm, acc);
    }
    #pragma unroll
    for (int rr = 0; rr < 8; ++rr)
      Og[base + (long)(qi * 16 + mb + rr) * kD + dj * 16 + r] = f2bf(acc[rr]);
  }
}

// ---------------- output projection (permutes back to [B,T,N,D] fp32) ----------------
__global__ __launch_bounds__(256) void outproj_wmma(
    const ushort_t* __restrict__ Ab,  // attn out, [B*N*T, D] bf16
    const ushort_t* __restrict__ Wb, const float* __restrict__ bias,
    float* __restrict__ out) {
  const int lane = threadIdx.x & 31;
  const int tile = blockIdx.x * 8 + (threadIdx.x >> 5);
  const int col  = (tile & 31) << 4;
  const int row0 = (tile >> 5) << 4;   // rows ordered (b, n, t)
  const int r = lane & 15;

  const ushort_t* arow = Ab + (long)row0 * kD;
  const ushort_t* wcol = Wb + (long)col * kD;
  v8f acc = {};
  #pragma unroll
  for (int k = 0; k < kD; k += 32)
    acc = wmma_bf16(load_a16(arow + k, kD, lane),
                    load_b16(wcol + k, kD, lane), acc);

  const int c = col + r;
  const float bc = bias[c];
  const int mb = (lane & 16) ? 8 : 0;
  const int t0 = row0 & (kT - 1);
  const int bn = row0 >> 6;
  const int n  = bn & (kN - 1);
  const int b  = bn >> 7;
  float* obase = out + (((long)b * kT + t0) * kN + n) * kD + c;  // [B,T,N,D]
  #pragma unroll
  for (int rr = 0; rr < 8; ++rr)
    obase[(long)(mb + rr) * kOutRow] = acc[rr] + bc;
}

// ---------------- launch ----------------

extern "C" void kernel_launch(void* const* d_in, const int* in_sizes, int n_in,
                              void* d_out, int out_size, void* d_ws, size_t ws_size,
                              hipStream_t stream) {
  (void)in_sizes; (void)n_in; (void)out_size; (void)ws_size;
  const float* X     = (const float*)d_in[0];
  const float* Wq    = (const float*)d_in[1];
  const float* bq    = (const float*)d_in[2];
  const float* gq    = (const float*)d_in[3];
  const float* betaq = (const float*)d_in[4];
  const float* mq    = (const float*)d_in[5];
  const float* vq    = (const float*)d_in[6];
  const float* Wk    = (const float*)d_in[7];
  const float* bk    = (const float*)d_in[8];
  const float* gk    = (const float*)d_in[9];
  const float* betak = (const float*)d_in[10];
  const float* mk    = (const float*)d_in[11];
  const float* vk    = (const float*)d_in[12];
  const float* Wv    = (const float*)d_in[13];
  const float* bv    = (const float*)d_in[14];
  const float* Wo    = (const float*)d_in[15];
  const float* bo    = (const float*)d_in[16];

  const long NX  = (long)kB * kT * kN * kD;        // 33,554,432
  const long NXP = (long)kB * kN * kTP * kD;       // padded X elements
  ushort_t* Xp  = (ushort_t*)d_ws;
  ushort_t* Qb  = Xp + NXP;
  ushort_t* Kb  = Qb + NX;
  ushort_t* Vb  = Kb + NX;
  ushort_t* Wqb = Vb + NX;
  ushort_t* Wkb = Wqb + (long)kKS * kD * kD;
  ushort_t* Wvb = Wkb + (long)kKS * kD * kD;
  ushort_t* Wob = Wvb + (long)kD * kD;
  // total workspace ~ 66MB + 3*64MB + ~4MB of weights (~263 MB)

  // 1) precision converts + causal zero-pad
  zero_pad<<<(kB * kN * 2 * kD) / 256, 256, 0, stream>>>(Xp);
  cvt_x_pad<<<(int)(NX / 256), 256, 0, stream>>>(X, Xp);
  cvt_convw<<<(kKS * kD * kD + 255) / 256, 256, 0, stream>>>(Wq, Wqb);
  cvt_convw<<<(kKS * kD * kD + 255) / 256, 256, 0, stream>>>(Wk, Wkb);
  cvt_f32_bf16<<<(kD * kD + 255) / 256, 256, 0, stream>>>(Wv, Wvb, kD * kD);
  cvt_f32_bf16<<<(kD * kD + 255) / 256, 256, 0, stream>>>(Wo, Wob, kD * kD);

  // 2) Q/K causal-conv + BN (131072 tiles, 8 waves/block)
  conv_bn_wmma<<<16384, 256, 0, stream>>>(Xp, Wqb, bq, gq, betaq, mq, vq, Qb);
  conv_bn_wmma<<<16384, 256, 0, stream>>>(Xp, Wkb, bk, gk, betak, mk, vk, Kb);

  // 3) V = X @ Wv^T + bv
  vproj_wmma<<<16384, 256, 0, stream>>>(Xp, Wvb, bv, Vb);

  // 4) per-(b,n,h) attention, output in-place over Qb
  attn_wmma<<<kB * kN * kH, 256, 0, stream>>>(Qb, Kb, Vb, Qb);

  // 5) out = attnOut @ Wo^T + bo, permuted to [B,T,N,D] fp32
  outproj_wmma<<<16384, 256, 0, stream>>>(Qb, Wob, bo, (float*)d_out);
}
